// SD_attn_Cross_16131897164210
// MI455X (gfx1250) — compile-verified, matrix operand order checked by reading, not run on hardware
//
#include <hip/hip_runtime.h>
#include <cstdint>
#include <cstddef>

// ---------------------------------------------------------------------------
// Types for CDNA5 WMMA (wave32): D(16x16,f32) = A(16x32,bf16) x B(32x16,bf16) + C
// ---------------------------------------------------------------------------
typedef __bf16 bf16_t;
typedef __attribute__((ext_vector_type(16))) __bf16 v16bf;
typedef __attribute__((ext_vector_type(8)))  float  v8f;

__device__ __forceinline__ v8f wmma_bf16(v16bf a, v16bf b, v8f c) {
  // (neg_a, A, neg_b, B, c_mod, C, reuse_a, reuse_b)
  return __builtin_amdgcn_wmma_f32_16x16x32_bf16(false, a, false, b, (short)0, c,
                                                 false, false);
}

// Fragment loader: lane l -> row (row0 + l%16), 16 contiguous bf16 at
// k0 + 16*(l/16).  32-byte load -> 2x b128 (ds or global).
__device__ __forceinline__ v16bf frag(const bf16_t* base, int stride, int row0,
                                      int k0, int lane) {
  const bf16_t* p = base + (row0 + (lane & 15)) * stride + k0 + ((lane & 16) ? 16 : 0);
  return *(const v16bf*)p;
}

// 64x64 = (64x512) x (512x64) GEMM into a 4x4 tile-block of accumulators.
// ks kept as a runtime loop (no unroll) so only 4 A-frags + 1 B-frag are live
// alongside the 128-VGPR accumulator block -> no spills.
__device__ __forceinline__ void gemm64(const bf16_t* __restrict__ A, int lda,
                                       const bf16_t* __restrict__ B, int ldb,
                                       v8f acc[4][4], int lane) {
#pragma unroll 1
  for (int ks = 0; ks < 16; ++ks) {
    v16bf a0 = frag(A, lda, 0,  ks * 32, lane);
    v16bf a1 = frag(A, lda, 16, ks * 32, lane);
    v16bf a2 = frag(A, lda, 32, ks * 32, lane);
    v16bf a3 = frag(A, lda, 48, ks * 32, lane);
#pragma unroll
    for (int nt = 0; nt < 4; ++nt) {
      v16bf b = frag(B, ldb, nt * 16, ks * 32, lane);
      acc[0][nt] = wmma_bf16(a0, b, acc[0][nt]);
      acc[1][nt] = wmma_bf16(a1, b, acc[1][nt]);
      acc[2][nt] = wmma_bf16(a2, b, acc[2][nt]);
      acc[3][nt] = wmma_bf16(a3, b, acc[3][nt]);
    }
  }
}

__device__ __forceinline__ void zero_acc(v8f acc[4][4]) {
#pragma unroll
  for (int mt = 0; mt < 4; ++mt)
#pragma unroll
    for (int nt = 0; nt < 4; ++nt)
      acc[mt][nt] = v8f{0.f, 0.f, 0.f, 0.f, 0.f, 0.f, 0.f, 0.f};
}

// ---------------------------------------------------------------------------
// Kernel 0: convert + transpose weights to bf16, N-major (row n = output chan)
// ---------------------------------------------------------------------------
__global__ __launch_bounds__(256) void prep_weights(
    const float* __restrict__ qv_w, const float* __restrict__ k_w,
    const float* __restrict__ proj_w, bf16_t* __restrict__ wq,
    bf16_t* __restrict__ wv, bf16_t* __restrict__ wk, bf16_t* __restrict__ wp) {
  int id = blockIdx.x * 256 + threadIdx.x;     // 0 .. 4*262144-1
  int m = id >> 18;
  int local = id & 262143;
  int n = local >> 9, k = local & 511;
  float v;
  bf16_t* dst;
  if (m == 0)      { v = qv_w[k * 1024 + n];       dst = wq; }
  else if (m == 1) { v = qv_w[k * 1024 + 512 + n]; dst = wv; }
  else if (m == 2) { v = k_w[k * 512 + n];         dst = wk; }
  else             { v = proj_w[k * 512 + n];      dst = wp; }
  dst[n * 512 + k] = (bf16_t)v;
}

// ---------------------------------------------------------------------------
// Kernel 1: per-window fused QKV projection + RoPE + masked softmax attention.
// grid = 1024 windows, block = 256 threads = 8 waves (one wave per head).
// LDS: [0,8K) cosT  [8K,16K) sinT  [16K,80K) X-stage(bf16 64x512)
//      [80K, 80K+8*24K) per-head {Q/P 8K, K 8K, Vt 8K}   total 272 KB
// ---------------------------------------------------------------------------
__global__ __launch_bounds__(256) void attn_kernel(
    const float* __restrict__ x, const float* __restrict__ cross,
    const float* __restrict__ qv_b, const float* __restrict__ k_b,
    const bf16_t* __restrict__ wq, const bf16_t* __restrict__ wv,
    const bf16_t* __restrict__ wk, bf16_t* __restrict__ attn_out) {
  extern __shared__ char smem[];
  float*  cosT = (float*)smem;
  float*  sinT = cosT + 2048;
  bf16_t* sX   = (bf16_t*)(smem + 16384);

  const int tid  = threadIdx.x;
  const int lane = tid & 31;
  const int wid  = tid >> 5;     // head id
  const int l16  = lane & 15;
  const int lhi  = lane >> 4;

  // Window decode: W = ((b*64 + wy*8 + wx) * 4) + (dh*2 + dw)
  const int W    = blockIdx.x;
  const int bimg = W >> 8;
  const int wy   = (W >> 5) & 7;
  const int wx   = (W >> 2) & 7;
  const int dh   = (W >> 1) & 1;
  const int dw   = W & 1;

  // window-token n (0..63) -> flat global token index (roll + window + dilate)
  auto gtok = [&](int n) -> int {
    int i  = ((n >> 3) << 1) + dh;
    int j  = ((n & 7) << 1) + dw;
    int y  = (wy * 16 + i + 4) & 127;
    int xx = (wx * 16 + j + 4) & 127;
    return (bimg * 128 + y) * 128 + xx;
  };
  // Shift-mask region code (reference's mask collapses to a y-only code)
  auto codeOf = [&](int n) -> int {
    int i = ((n >> 3) << 1) + dh;
    int y = wy * 16 + i;
    return (y < 120) ? 0 : ((y < 124) ? 1 : 2);
  };

  // --- Phase A: RoPE tables + stage X window into LDS (fp32 -> bf16) -------
  for (int idx = tid; idx < 2048; idx += 256) {
    int n = idx >> 5, p = idx & 31;
    int t = p & 15;
    float inv = __expf(-9.210340372f * (float)(2 * t) * (1.0f / 32.0f));
    float pos = (p < 16) ? (float)(n >> 3) : (float)(n & 7);
    float ang = pos * inv;
    cosT[idx] = __cosf(ang);
    sinT[idx] = __sinf(ang);
  }
  for (int idx = tid; idx < 64 * 128; idx += 256) {
    int row = idx >> 7;
    int c4  = idx & 127;
    float4 f = *(const float4*)(x + (size_t)gtok(row) * 512 + c4 * 4);
    bf16_t* d = sX + row * 512 + c4 * 4;
    d[0] = (bf16_t)f.x; d[1] = (bf16_t)f.y; d[2] = (bf16_t)f.z; d[3] = (bf16_t)f.w;
  }
  __syncthreads();

  bf16_t* sQ  = (bf16_t*)(smem + 16384 + 65536 + wid * 24576);  // 64x64, reused as P
  bf16_t* sK  = sQ + 4096;                                      // 64x64 row-major
  bf16_t* sVt = sQ + 8192;                                      // 64x64 transposed

  const float scale = 0.125f;  // HD^-0.5
  v8f acc[4][4];

  // --- Q = Xw @ Wq_h  (+bias, RoPE) ---------------------------------------
  zero_acc(acc);
  gemm64(sX, 512, wq + (size_t)(wid * 64) * 512, 512, acc, lane);
#pragma unroll
  for (int nt = 0; nt < 4; ++nt) {
    float bias = qv_b[wid * 64 + nt * 16 + l16];
    int colg = nt * 16 + l16;
    int p = colg >> 1;
    int odd = colg & 1;
#pragma unroll
    for (int mt = 0; mt < 4; ++mt) {
#pragma unroll
      for (int i = 0; i < 8; ++i) {
        int n = mt * 16 + i + lhi * 8;
        float v0 = acc[mt][nt][i] + bias;
        float other = __shfl_xor(v0, 1, 32);  // RoPE pair partner (adjacent col)
        float c = cosT[n * 32 + p], s = sinT[n * 32 + p];
        float r = odd ? (other * s + v0 * c) : (v0 * c - other * s);
        sQ[n * 64 + colg] = (bf16_t)r;
      }
    }
  }

  // --- V = Xw @ Wv_h (+bias), stored transposed for B-fragments -----------
  zero_acc(acc);
  gemm64(sX, 512, wv + (size_t)(wid * 64) * 512, 512, acc, lane);
#pragma unroll
  for (int nt = 0; nt < 4; ++nt) {
    float bias = qv_b[512 + wid * 64 + nt * 16 + l16];
#pragma unroll
    for (int mt = 0; mt < 4; ++mt) {
#pragma unroll
      for (int i = 0; i < 8; ++i) {
        int n = mt * 16 + i + lhi * 8;
        sVt[(nt * 16 + l16) * 64 + n] = (bf16_t)(acc[mt][nt][i] + bias);
      }
    }
  }

  __syncthreads();  // done reading X; re-stage with cross
  for (int idx = tid; idx < 64 * 128; idx += 256) {
    int row = idx >> 7;
    int c4  = idx & 127;
    float4 f = *(const float4*)(cross + (size_t)gtok(row) * 512 + c4 * 4);
    bf16_t* d = sX + row * 512 + c4 * 4;
    d[0] = (bf16_t)f.x; d[1] = (bf16_t)f.y; d[2] = (bf16_t)f.z; d[3] = (bf16_t)f.w;
  }
  __syncthreads();

  // --- K = Cw @ Wk_h (+bias, RoPE) ----------------------------------------
  zero_acc(acc);
  gemm64(sX, 512, wk + (size_t)(wid * 64) * 512, 512, acc, lane);
#pragma unroll
  for (int nt = 0; nt < 4; ++nt) {
    float bias = k_b[wid * 64 + nt * 16 + l16];
    int colg = nt * 16 + l16;
    int p = colg >> 1;
    int odd = colg & 1;
#pragma unroll
    for (int mt = 0; mt < 4; ++mt) {
#pragma unroll
      for (int i = 0; i < 8; ++i) {
        int n = mt * 16 + i + lhi * 8;
        float v0 = acc[mt][nt][i] + bias;
        float other = __shfl_xor(v0, 1, 32);
        float c = cosT[n * 32 + p], s = sinT[n * 32 + p];
        float r = odd ? (other * s + v0 * c) : (v0 * c - other * s);
        sK[n * 64 + colg] = (bf16_t)r;
      }
    }
  }

  // --- Attention: S = Q K^T, softmax(+mask) in-register, O = P V -----------
#pragma unroll 1
  for (int mt = 0; mt < 4; ++mt) {
    v8f accS[4];
#pragma unroll
    for (int nt = 0; nt < 4; ++nt) accS[nt] = v8f{0.f, 0.f, 0.f, 0.f, 0.f, 0.f, 0.f, 0.f};
#pragma unroll
    for (int ks = 0; ks < 2; ++ks) {
      v16bf a = frag(sQ, 64, mt * 16, ks * 32, lane);
#pragma unroll
      for (int nt = 0; nt < 4; ++nt) {
        v16bf b = frag(sK, 64, nt * 16, ks * 32, lane);  // B col m' = K row m'
        accS[nt] = wmma_bf16(a, b, accS[nt]);
      }
    }
    // row softmax: each row lives in one 16-lane group across 4 nt tiles
#pragma unroll
    for (int i = 0; i < 8; ++i) {
      int n  = mt * 16 + i + lhi * 8;
      int cq = codeOf(n);
      float vals[4];
#pragma unroll
      for (int nt = 0; nt < 4; ++nt) {
        int m  = nt * 16 + l16;
        int ck = codeOf(m);
        vals[nt] = accS[nt][i] * scale + ((ck == cq) ? 0.f : -1e9f);
      }
      float mx = fmaxf(fmaxf(vals[0], vals[1]), fmaxf(vals[2], vals[3]));
#pragma unroll
      for (int s = 1; s < 16; s <<= 1) mx = fmaxf(mx, __shfl_xor(mx, s, 32));
      float sum = 0.f;
#pragma unroll
      for (int nt = 0; nt < 4; ++nt) { vals[nt] = __expf(vals[nt] - mx); sum += vals[nt]; }
#pragma unroll
      for (int s = 1; s < 16; s <<= 1) sum += __shfl_xor(sum, s, 32);
      float rs = 1.f / sum;
#pragma unroll
      for (int nt = 0; nt < 4; ++nt)
        sQ[n * 64 + nt * 16 + l16] = (bf16_t)(vals[nt] * rs);  // P overwrites Q rows
    }
    // O rows = P rows @ V
    v8f accO[4];
#pragma unroll
    for (int nt = 0; nt < 4; ++nt) accO[nt] = v8f{0.f, 0.f, 0.f, 0.f, 0.f, 0.f, 0.f, 0.f};
#pragma unroll
    for (int ks = 0; ks < 2; ++ks) {
      v16bf a = frag(sQ, 64, mt * 16, ks * 32, lane);
#pragma unroll
      for (int nt = 0; nt < 4; ++nt) {
        v16bf b = frag(sVt, 64, nt * 16, ks * 32, lane);  // B col ch = Vt row ch
        accO[nt] = wmma_bf16(a, b, accO[nt]);
      }
    }
#pragma unroll
    for (int nt = 0; nt < 4; ++nt) {
#pragma unroll
      for (int i = 0; i < 8; ++i) {
        int n = mt * 16 + i + lhi * 8;
        attn_out[(size_t)gtok(n) * 512 + wid * 64 + nt * 16 + l16] = (bf16_t)accO[nt][i];
      }
    }
  }
}

// ---------------------------------------------------------------------------
// Kernel 2: out = attn(bf16, token-major) @ proj_w + proj_b (fp32 out)
// block computes 128 rows x 256 cols; 8 waves = 2x4 grid of 64x64 tiles.
// ---------------------------------------------------------------------------
__global__ __launch_bounds__(256) void proj_kernel(
    const bf16_t* __restrict__ attn, const bf16_t* __restrict__ wp,
    const float* __restrict__ proj_b, float* __restrict__ out) {
  extern __shared__ char smem[];
  bf16_t* sA = (bf16_t*)smem;  // 128 x 512 bf16 = 128 KB
  const int tid = threadIdx.x, lane = tid & 31, wid = tid >> 5;
  const int l16 = lane & 15, lhi = lane >> 4;
  const size_t row0 = (size_t)blockIdx.x * 128;

  const uint4* src = (const uint4*)(attn + row0 * 512);
  uint4* dst = (uint4*)sA;
  for (int i = tid; i < 8192; i += 256) dst[i] = src[i];
  __syncthreads();

  const int wm = wid >> 2, wn = wid & 3;
  const int cb = blockIdx.y * 256 + wn * 64;  // global col base
  const int rb = wm * 64;                     // row base within block

  v8f acc[4][4];
  zero_acc(acc);
  gemm64(sA + rb * 512, 512, wp + (size_t)cb * 512, 512, acc, lane);

#pragma unroll
  for (int nt = 0; nt < 4; ++nt) {
    float bias = proj_b[cb + nt * 16 + l16];
#pragma unroll
    for (int mt = 0; mt < 4; ++mt) {
#pragma unroll
      for (int i = 0; i < 8; ++i) {
        size_t tok = row0 + rb + mt * 16 + i + lhi * 8;
        out[tok * 512 + cb + nt * 16 + l16] = acc[mt][nt][i] + bias;
      }
    }
  }
}

// ---------------------------------------------------------------------------
extern "C" void kernel_launch(void* const* d_in, const int* in_sizes, int n_in,
                              void* d_out, int out_size, void* d_ws, size_t ws_size,
                              hipStream_t stream) {
  const float* x      = (const float*)d_in[0];
  const float* cross  = (const float*)d_in[1];
  const float* qv_w   = (const float*)d_in[2];
  const float* qv_b   = (const float*)d_in[3];
  const float* k_w    = (const float*)d_in[4];
  const float* k_b    = (const float*)d_in[5];
  const float* proj_w = (const float*)d_in[6];
  const float* proj_b = (const float*)d_in[7];
  float* out = (float*)d_out;

  char* ws = (char*)d_ws;
  bf16_t* wq   = (bf16_t*)(ws + 0 * 524288);
  bf16_t* wv   = (bf16_t*)(ws + 1 * 524288);
  bf16_t* wk   = (bf16_t*)(ws + 2 * 524288);
  bf16_t* wp   = (bf16_t*)(ws + 3 * 524288);
  bf16_t* attn = (bf16_t*)(ws + 4 * 524288);  // 65536 x 512 bf16 = 64 MB

  const int SMEM1 = 16384 + 65536 + 8 * 24576;  // 278528 B (<= 320 KB WGP LDS)
  const int SMEM2 = 128 * 512 * 2;              // 131072 B
  hipFuncSetAttribute(reinterpret_cast<const void*>(attn_kernel),
                      hipFuncAttributeMaxDynamicSharedMemorySize, SMEM1);
  hipFuncSetAttribute(reinterpret_cast<const void*>(proj_kernel),
                      hipFuncAttributeMaxDynamicSharedMemorySize, SMEM2);

  prep_weights<<<4096, 256, 0, stream>>>(qv_w, k_w, proj_w, wq, wv, wk, wp);
  attn_kernel<<<1024, 256, SMEM1, stream>>>(x, cross, qv_b, k_b, wq, wv, wk, attn);
  proj_kernel<<<dim3(512, 2), 256, SMEM2, stream>>>(attn, wp, proj_b, out);
}
